// GAT_4561255268883
// MI455X (gfx1250) — matrix operand order repaired and checked by reading, not compiled, not run on hardware
//
#include <hip/hip_runtime.h>

typedef __attribute__((ext_vector_type(16))) _Float16 v16h;
typedef __attribute__((ext_vector_type(8)))  float    v8f;

#define Bn 8
#define Nn 2048
#define Cc 128
#define Ff 128
#define ALPHA 0.2f

// ---------------------------------------------------------------------------
// K1: h = text @ W  (f16 WMMA, f32 accumulate). One wave per 16-row tile.
// W is transposed into LDS as f16 (WT[f][c]) so B-matrix reads are contiguous.
// ---------------------------------------------------------------------------
__global__ __launch_bounds__(128) void k1_gemm_h(const float* __restrict__ text,
                                                 const float* __restrict__ W,
                                                 float* __restrict__ h) {
    __shared__ __align__(32) _Float16 WT[Ff * Cc];   // 32 KB
    const int tid = threadIdx.x;
    for (int i = tid; i < Ff * Cc; i += 128) {
        int f = i >> 7, c = i & 127;                 // WT[f][c] = W[c][f]
        WT[i] = (_Float16)W[c * Ff + f];
    }
    __syncthreads();

    const int wave = tid >> 5;
    const int lane = tid & 31;
    const int half = lane >> 4;
    const int m    = lane & 15;                      // A row / B column
    const long tile = (long)blockIdx.x * 4 + wave;   // global 16-row tile
    const int  b    = (int)(tile >> 7);              // 128 tiles per batch
    const int  i0   = (int)(tile & 127) << 4;

    const float* tb = text + ((long)b * Nn + i0) * Cc;

    v8f acc[8];
#pragma unroll
    for (int t = 0; t < 8; ++t)
#pragma unroll
        for (int r = 0; r < 8; ++r) acc[t][r] = 0.0f;

#pragma unroll
    for (int kc = 0; kc < Cc; kc += 32) {
        // A tile: text rows, K = 16*half + e
        const float* rp = tb + (long)m * Cc + kc + 16 * half;
        float4 g0 = *(const float4*)(rp);
        float4 g1 = *(const float4*)(rp + 4);
        float4 g2 = *(const float4*)(rp + 8);
        float4 g3 = *(const float4*)(rp + 12);
        v16h A;
        A[0]=(_Float16)g0.x; A[1]=(_Float16)g0.y; A[2]=(_Float16)g0.z; A[3]=(_Float16)g0.w;
        A[4]=(_Float16)g1.x; A[5]=(_Float16)g1.y; A[6]=(_Float16)g1.z; A[7]=(_Float16)g1.w;
        A[8]=(_Float16)g2.x; A[9]=(_Float16)g2.y; A[10]=(_Float16)g2.z; A[11]=(_Float16)g2.w;
        A[12]=(_Float16)g3.x; A[13]=(_Float16)g3.y; A[14]=(_Float16)g3.z; A[15]=(_Float16)g3.w;

#pragma unroll
        for (int t = 0; t < 8; ++t) {
            // B element e -> WT[(16t+n)][kc + 16*half + e]  (contiguous, 32B aligned)
            const v16h Bv = *(const v16h*)&WT[((t << 4) + m) * Cc + kc + 16 * half];
            acc[t] = __builtin_amdgcn_wmma_f32_16x16x32_f16(
                false, A, false, Bv, (short)0, acc[t], false, false);
        }
    }

    float* hb = h + ((long)b * Nn + i0) * Ff;
#pragma unroll
    for (int t = 0; t < 8; ++t)
#pragma unroll
        for (int r = 0; r < 8; ++r)
            hb[(long)(r + 8 * half) * Ff + (t << 4) + m] = acc[t][r];
}

// ---------------------------------------------------------------------------
// K2: per-row e1 = h·a1, e2 = h·a2, plus the factorized softmax weights
//     uv[row] = (exp(e2), exp(ALPHA*e2)).  One wave per row.
// ---------------------------------------------------------------------------
__global__ __launch_bounds__(256) void k2_e(const float* __restrict__ h,
                                            const float* __restrict__ a,
                                            float* __restrict__ e1,
                                            float* __restrict__ e2,
                                            float2* __restrict__ uv) {
    const int lane = threadIdx.x & 31;
    const long row = (long)blockIdx.x * 8 + (threadIdx.x >> 5);
    const float* hr = h + row * Ff;
    float4 hv = *(const float4*)(hr + lane * 4);
    float4 a1 = *(const float4*)(a + lane * 4);
    float4 a2 = *(const float4*)(a + Ff + lane * 4);
    float s1 = hv.x*a1.x + hv.y*a1.y + hv.z*a1.z + hv.w*a1.w;
    float s2 = hv.x*a2.x + hv.y*a2.y + hv.z*a2.z + hv.w*a2.w;
#pragma unroll
    for (int off = 16; off >= 1; off >>= 1) {
        s1 += __shfl_xor(s1, off, 32);
        s2 += __shfl_xor(s2, off, 32);
    }
    if (lane == 0) {
        e1[row] = s1;
        e2[row] = s2;
        float2 p; p.x = __expf(s2); p.y = __expf(ALPHA * s2);
        uv[row] = p;
    }
}

// ---------------------------------------------------------------------------
// K3: hT[b][f][n] = (f16) h[b][n][f]  (B-matrix operand for the attention GEMM)
// ---------------------------------------------------------------------------
__global__ __launch_bounds__(256) void k3_tr(const float* __restrict__ h,
                                             _Float16* __restrict__ hT) {
    const long idx = (long)blockIdx.x * 256 + threadIdx.x;   // n fastest
    const int  n  = (int)(idx & (Nn - 1));
    const long bf = idx >> 11;
    const int  f  = (int)(bf & (Ff - 1));
    const long b  = bf >> 7;
    hT[idx] = (_Float16)h[(b * Nn + n) * Ff + f];
}

// ---------------------------------------------------------------------------
// K4: softmax(leaky_relu(e1_i + e2_j)) @ h with fused elu epilogue.
// Exact row max = lrelu(e1_i + max_j e2_j)  (lrelu is monotone), and
// exp(lrelu(x) - m) = max(c1*u_j, c2*v_j) with per-row c1,c2 -> the inner
// loop has ZERO transcendentals (keeps the TRANS pipe free for XDL WMMA).
// One wave per 16-row tile; j streamed in chunks of 32; (u,v) pairs in LDS.
// ---------------------------------------------------------------------------
__global__ __launch_bounds__(128) void k4_attn(const float* __restrict__ h,
                                               const _Float16* __restrict__ hT,
                                               const float* __restrict__ e1g,
                                               const float* __restrict__ e2g,
                                               const float2* __restrict__ uvg,
                                               float* __restrict__ out) {
    __shared__ __align__(16) float uvs[2 * Nn];              // 16 KB (u,v pairs)
    __shared__ float wmax[4];
    const int tid  = threadIdx.x;
    const int wave = tid >> 5;
    const int lane = tid & 31;
    const int half = lane >> 4;
    const int m    = lane & 15;

    const long tile0 = (long)blockIdx.x * 4;
    const int  b     = (int)(tile0 >> 7);                    // 4 | 128 -> same batch
    const int  i0    = (int)((tile0 + wave) & 127) << 4;

    // Cooperative LDS fill of (u,v) pairs + per-batch max(e2) reduction.
    const float4* uvg4 = (const float4*)(uvg + (long)b * Nn);
    for (int i = tid; i < Nn / 2; i += 128) ((float4*)uvs)[i] = uvg4[i];
    float pm = -__builtin_inff();
    const float4* e2g4 = (const float4*)(e2g + (long)b * Nn);
    for (int i = tid; i < Nn / 4; i += 128) {
        float4 v = e2g4[i];
        pm = fmaxf(pm, fmaxf(fmaxf(v.x, v.y), fmaxf(v.z, v.w)));
    }
#pragma unroll
    for (int off = 16; off >= 1; off >>= 1) pm = fmaxf(pm, __shfl_xor(pm, off, 32));
    if (lane == 0) wmax[wave] = pm;
    __syncthreads();
    const float e2max = fmaxf(fmaxf(wmax[0], wmax[1]), fmaxf(wmax[2], wmax[3]));

    // Per-row constants: exact softmax max m, factor constants c1,c2.
    const float e1v = e1g[(long)b * Nn + i0 + m];
    const float xm  = e1v + e2max;
    const float mrow = fmaxf(xm, ALPHA * xm);                // lrelu(e1+max e2)
    const float c1 = __expf(e1v - mrow);
    const float c2 = __expf(ALPHA * e1v - mrow);

    float Zl = 0.0f;                                         // per-lane partial row sum
    v8f acc[8];
#pragma unroll
    for (int t = 0; t < 8; ++t)
#pragma unroll
        for (int r = 0; r < 8; ++r) acc[t][r] = 0.0f;

    const _Float16* hTb = hT + (long)b * Ff * Nn;

    for (int j0 = 0; j0 < Nn; j0 += 32) {
        // A-layout: element e -> K = 16*half + e
        const float2* up = (const float2*)uvs + j0 + 16 * half;
        float pv[16];
#pragma unroll
        for (int e = 0; e < 16; ++e) {
            float2 w = up[e];
            float p = fmaxf(c1 * w.x, c2 * w.y);             // exp(lrelu(x)-m)
            Zl += p;
            pv[e] = p;
        }
        v16h A;
#pragma unroll
        for (int e = 0; e < 16; ++e) A[e] = (_Float16)pv[e];

#pragma unroll
        for (int t = 0; t < 8; ++t) {
            // B element e -> hT[(16t+n)][j0 + 16*half + e] (contiguous, 32B aligned)
            const v16h Bv = *(const v16h*)(hTb + (long)((t << 4) + m) * Nn + j0 + 16 * half);
            acc[t] = __builtin_amdgcn_wmma_f32_16x16x32_f16(
                false, A, false, Bv, (short)0, acc[t], false, false);
        }
    }

    // Row sums: combine the two half-lanes once, then broadcast per acc row.
    const float Zrow = Zl + __shfl_xor(Zl, 16, 32);
    float invZ[8];
#pragma unroll
    for (int r = 0; r < 8; ++r) invZ[r] = 1.0f / __shfl(Zrow, 8 * half + r, 32);

    const float* hb = h   + ((long)b * Nn + i0) * Ff;
    float*       ob = out + ((long)b * Nn + i0) * Ff;
#pragma unroll
    for (int t = 0; t < 8; ++t)
#pragma unroll
        for (int r = 0; r < 8; ++r) {
            long off = (long)(r + 8 * half) * Ff + (t << 4) + m;
            float x = acc[t][r] * invZ[r] + ALPHA * hb[off];
            ob[off] = (x > 0.0f) ? x : (__expf(x) - 1.0f);   // elu (alpha=1)
        }
}

// ---------------------------------------------------------------------------
extern "C" void kernel_launch(void* const* d_in, const int* in_sizes, int n_in,
                              void* d_out, int out_size, void* d_ws, size_t ws_size,
                              hipStream_t stream) {
    const float* text = (const float*)d_in[0];
    // d_in[1] (adj) is unused by the reference — never read (saves 128 MB of traffic)
    const float* W = (const float*)d_in[2];
    const float* a = (const float*)d_in[3];
    float* out = (float*)d_out;

    char* ws = (char*)d_ws;
    float*     h  = (float*)ws;                                    //  8 MB f32
    _Float16*  hT = (_Float16*)(ws + (size_t)Bn * Nn * Ff * 4);    //  4 MB f16
    float*     e1 = (float*)(ws + (size_t)Bn * Nn * Ff * 6);       // 64 KB
    float*     e2 = e1 + (size_t)Bn * Nn;                          // 64 KB
    float2*    uv = (float2*)(e2 + (size_t)Bn * Nn);               // 128 KB

    k1_gemm_h<<<Bn * Nn / 64, 128, 0, stream>>>(text, W, h);
    k2_e     <<<Bn * Nn / 8, 256, 0, stream>>>(h, a, e1, e2, uv);
    k3_tr    <<<(size_t)Bn * Ff * Nn / 256, 256, 0, stream>>>(h, hT);
    k4_attn  <<<Bn * Nn / 64, 128, 0, stream>>>(h, hT, e1, e2, uv, out);
}
